// LinearGlobalAttention_25228637897370
// MI455X (gfx1250) — matrix operand order repaired and checked by reading, not compiled, not run on hardware
//
#include <hip/hip_runtime.h>
#include <cstdint>

// ---------------------------------------------------------------------------
// Types
// ---------------------------------------------------------------------------
typedef __attribute__((ext_vector_type(16))) __bf16 bf16x16;
typedef __attribute__((ext_vector_type(8)))  __bf16 bf16x8;
typedef __attribute__((ext_vector_type(4)))  __bf16 bf16x4;
typedef __attribute__((ext_vector_type(2)))  __bf16 bf16x2;
typedef __attribute__((ext_vector_type(8)))  float  f32x8;

#define DEVINL __device__ __attribute__((always_inline)) inline

static constexpr int CDIM = 256;
static constexpr int NROW = 262144;

// ---------------------------------------------------------------------------
// Optional CDNA5 async global->LDS copy (pure-copy staging path)
// Builtin signature (probed from diagnostics): 
//   (int4 addrspace(1)* gsrc, int4 addrspace(3)* ldsdst, imm int off, imm int cpol)
// ---------------------------------------------------------------------------
#if defined(__has_builtin)
#if __has_builtin(__builtin_amdgcn_global_load_async_to_lds_b128) && \
    __has_builtin(__builtin_amdgcn_s_wait_asynccnt)
#define USE_ASYNC_LDS 1
#endif
#endif

#ifdef USE_ASYNC_LDS
typedef __attribute__((__vector_size__(16))) int v4i_;
typedef __attribute__((address_space(1))) v4i_ g_v4i;
typedef __attribute__((address_space(3))) v4i_ l_v4i;
DEVINL void async_copy_b128(const void* g, void* l) {
    // generic LDS address truncates to the LDS byte offset (flat aperture rule);
    // as3 pointers are 32-bit on amdgcn.
    __builtin_amdgcn_global_load_async_to_lds_b128(
        (g_v4i*)(uintptr_t)g, (l_v4i*)(uint32_t)(uintptr_t)l, 0, 0);
}
DEVINL void async_wait0() { __builtin_amdgcn_s_wait_asynccnt(0); }
#endif

// ---------------------------------------------------------------------------
// WMMA fragment helpers (CDNA5 16x16x32 bf16 layouts, wave32)
// A 16x32: lane<16 -> M=lane, K elems {k0..k0+7, k0+16..k0+23};
//          lane>=16 -> same M, K elems shifted by +8.
// ---------------------------------------------------------------------------
DEVINL bf16x16 frag_a(const __bf16* p, int rs) {
    const int lane = threadIdx.x & 31;
    const __bf16* q = p + (lane & 15) * rs + ((lane >> 4) << 3);
    bf16x8 lo = *(const bf16x8*)(q);
    bf16x8 hi = *(const bf16x8*)(q + 16);
    return __builtin_shufflevector(lo, hi, 0,1,2,3,4,5,6,7,8,9,10,11,12,13,14,15);
}

// B 32x16: lane<16 -> N=lane, K elems {k0..k0+7, k0+16..k0+23}; lane>=16 -> K shifted by +8.
DEVINL bf16x16 frag_b(const __bf16* p, int rs) {
    const int lane = threadIdx.x & 31;
    const __bf16* q = p + (lane & 15) * rs + ((lane >> 4) << 3);
    bf16x8 lo = *(const bf16x8*)(q);
    bf16x8 hi = *(const bf16x8*)(q + 16);
    return __builtin_shufflevector(lo, hi, 0,1,2,3,4,5,6,7,8,9,10,11,12,13,14,15);
}

DEVINL f32x8 wmma_bf16(bf16x16 a, bf16x16 b, f32x8 c) {
    return __builtin_amdgcn_wmma_f32_16x16x32_bf16(false, a, false, b, (short)0, c,
                                                   false, false);
}

DEVINL float elu1(float x) { return x > 0.0f ? x + 1.0f : __expf(x); }

// ---------------------------------------------------------------------------
// K0: f32 -> bf16 weight conversion
// ---------------------------------------------------------------------------
__global__ void k_cvt(const float* __restrict__ s, __bf16* __restrict__ d, int n) {
    int i = blockIdx.x * blockDim.x + threadIdx.x;
    if (i < n) d[i] = (__bf16)s[i];
}

// ---------------------------------------------------------------------------
// K1: q = elu(x @ Wq^T) + 1   (bf16 out).  128 rows per WG, 8 waves.
// smem: xs bf16[128][264] @0 (67584B), qstage bf16[128][72] @67584 (18432B)
// ---------------------------------------------------------------------------
__global__ void k_qproj(const float* __restrict__ x, const __bf16* __restrict__ Wq,
                        __bf16* __restrict__ qout) {
    extern __shared__ char smem[];
    __bf16* xs     = (__bf16*)(smem);
    __bf16* qstage = (__bf16*)(smem + 67584);
    const int t    = threadIdx.x;
    const int lane = t & 31, w = t >> 5;
    const int half = lane >> 4, ln = lane & 15;
    const size_t row0 = (size_t)blockIdx.x * 128;

    // load x tile -> bf16 LDS
    const float4* xv = (const float4*)(x + row0 * CDIM);
#pragma unroll
    for (int i = 0; i < 32; ++i) {
        int idx = t + i * 256;
        int r = idx >> 6, c4 = idx & 63;
        float4 v = xv[r * 64 + c4];
        bf16x4 b;
        b[0] = (__bf16)v.x; b[1] = (__bf16)v.y; b[2] = (__bf16)v.z; b[3] = (__bf16)v.w;
        *(bf16x4*)(xs + r * 264 + c4 * 4) = b;
    }
    __syncthreads();

    const int m = w;  // M-tile per wave
    bf16x16 af[8];
#pragma unroll
    for (int kk = 0; kk < 8; ++kk) af[kk] = frag_a(xs + m * 16 * 264 + kk * 32, 264);

    for (int c = 0; c < 4; ++c) {  // 4 chunks of 64 output cols
#pragma unroll
        for (int nt = 0; nt < 4; ++nt) {
            int n0 = c * 64 + nt * 16;
            f32x8 acc = {0.f,0.f,0.f,0.f,0.f,0.f,0.f,0.f};
#pragma unroll
            for (int kk = 0; kk < 8; ++kk)
                acc = wmma_bf16(af[kk], frag_b(Wq + n0 * CDIM + kk * 32, CDIM), acc);
#pragma unroll
            for (int j = 0; j < 8; ++j) {
                int r = m * 16 + half * 8 + j;
                qstage[r * 72 + nt * 16 + ln] = (__bf16)elu1(acc[j]);
            }
        }
        __syncthreads();
#pragma unroll
        for (int i = 0; i < 8; ++i) {  // coalesced store 128x64 bf16
            int idx = t + i * 256;
            int r = idx >> 4, c4 = idx & 15;
            bf16x4 v = *(bf16x4*)(qstage + r * 72 + c4 * 4);
            *(bf16x4*)(qout + (row0 + r) * CDIM + c * 64 + c4 * 4) = v;
        }
        __syncthreads();
    }
}

// ---------------------------------------------------------------------------
// K2: recompute k,v per 128-row chunk; accumulate kv = k^T v in registers,
// ksum partials in fixed order.  256 WGs x 1024 rows, 1024 threads (32 waves).
// smem: xs bf16[128][264] @0, ktT bf16[256][136] @67584, vtT @137216,
//       ksp f32[768] @206848  -> total 209920 B
// ---------------------------------------------------------------------------
__global__ void k_kv(const float* __restrict__ x, const __bf16* __restrict__ Wk,
                     const __bf16* __restrict__ Wv, float* __restrict__ kvpart,
                     float* __restrict__ kspart) {
    extern __shared__ char smem[];
    __bf16* xs  = (__bf16*)(smem);
    __bf16* ktT = (__bf16*)(smem + 67584);
    __bf16* vtT = (__bf16*)(smem + 137216);
    float*  ksp = (float*)(smem + 206848);

    const int t    = threadIdx.x;
    const int lane = t & 31, w = t >> 5;
    const int half = lane >> 4, ln = lane & 15;
    const int wg   = blockIdx.x;

    // projection role: sel 0=k / 1=v, M-tile m, N-half nh
    const int sel = w & 1;
    const int m   = (w >> 1) & 7;
    const int nh  = w >> 4;
    const __bf16* W = sel ? Wv : Wk;
    __bf16* T       = sel ? vtT : ktT;

    // kv role: a-tile per wave pair, b-half
    const int at = w >> 1;
    const int bh = w & 1;

    f32x8 kvacc[8];
#pragma unroll
    for (int i = 0; i < 8; ++i) kvacc[i] = (f32x8){0.f,0.f,0.f,0.f,0.f,0.f,0.f,0.f};
    float ksr = 0.0f;
    const int kcol = t & 255, kq = t >> 8;  // ksum: column + quarter

    for (int chunk = 0; chunk < 8; ++chunk) {
        const size_t row0 = (size_t)wg * 1024 + (size_t)chunk * 128;
        // load x chunk -> bf16 LDS
        const float4* xv = (const float4*)(x + row0 * CDIM);
#pragma unroll
        for (int i = 0; i < 8; ++i) {
            int idx = t + i * 1024;
            int r = idx >> 6, c4 = idx & 63;
            float4 v = xv[r * 64 + c4];
            bf16x4 b;
            b[0] = (__bf16)v.x; b[1] = (__bf16)v.y; b[2] = (__bf16)v.z; b[3] = (__bf16)v.w;
            *(bf16x4*)(xs + r * 264 + c4 * 4) = b;
        }
        __syncthreads();

        // k / v projection, store transposed tiles (packed b32 stores)
        {
            bf16x16 af[8];
#pragma unroll
            for (int kk = 0; kk < 8; ++kk)
                af[kk] = frag_a(xs + m * 16 * 264 + kk * 32, 264);
#pragma unroll
            for (int i = 0; i < 8; ++i) {
                int nt = nh * 8 + i;
                f32x8 acc = {0.f,0.f,0.f,0.f,0.f,0.f,0.f,0.f};
#pragma unroll
                for (int kk = 0; kk < 8; ++kk)
                    acc = wmma_bf16(af[kk], frag_b(W + nt * 16 * CDIM + kk * 32, CDIM), acc);
                int a  = nt * 16 + ln;          // feature index (transposed row)
                int r0 = m * 16 + half * 8;     // data row base (transposed col)
#pragma unroll
                for (int j = 0; j < 8; j += 2) {  // consecutive j -> consecutive r
                    float v0 = acc[j], v1 = acc[j + 1];
                    if (!sel) { v0 = elu1(v0); v1 = elu1(v1); }
                    bf16x2 pr;
                    pr[0] = (__bf16)v0; pr[1] = (__bf16)v1;
                    *(bf16x2*)(T + a * 136 + r0 + j) = pr;
                }
            }
        }
        __syncthreads();

        // ksum partial accumulation (fixed order, from ktT)
        {
            const __bf16* kp = ktT + kcol * 136 + kq * 32;
#pragma unroll
            for (int r = 0; r < 32; r += 8) {
                bf16x8 k8 = *(const bf16x8*)(kp + r);
#pragma unroll
                for (int e = 0; e < 8; ++e) ksr += (float)k8[e];
            }
        }

        // kv += k^T v over this chunk (K = 128, 4 WMMA k-steps)
#pragma unroll
        for (int ks = 0; ks < 4; ++ks) {
            bf16x16 a2 = frag_a(ktT + at * 16 * 136 + ks * 32, 136);
#pragma unroll
            for (int i = 0; i < 8; ++i) {
                int bt = bh * 8 + i;
                kvacc[i] = wmma_bf16(a2, frag_b(vtT + bt * 16 * 136 + ks * 32, 136),
                                     kvacc[i]);
            }
        }
        __syncthreads();
    }

    // write kv partial
    float* outp = kvpart + (size_t)wg * 65536;
#pragma unroll
    for (int i = 0; i < 8; ++i) {
        int bt = bh * 8 + i;
#pragma unroll
        for (int j = 0; j < 8; ++j) {
            int a = at * 16 + half * 8 + j;
            int b = bt * 16 + ln;
            outp[a * 256 + b] = kvacc[i][j];
        }
    }
    // ksum partial: deterministic 4-way combine via LDS
    if (kq > 0) ksp[(kq - 1) * 256 + kcol] = ksr;
    __syncthreads();
    if (kq == 0)
        kspart[(size_t)wg * 256 + kcol] = ksr + ksp[kcol] + ksp[256 + kcol] + ksp[512 + kcol];
}

// ---------------------------------------------------------------------------
// K3a: reduce 256 kv/ksum partials (fixed order)
// ---------------------------------------------------------------------------
__global__ void k_reduce(const float* __restrict__ kvpart, const float* __restrict__ kspart,
                         __bf16* __restrict__ kvb, float* __restrict__ ksum) {
    const int a = blockIdx.x, b = threadIdx.x;
    float acc = 0.0f;
#pragma unroll 4
    for (int p = 0; p < 256; ++p) acc += kvpart[(size_t)p * 65536 + a * 256 + b];
    kvb[a * 256 + b] = (__bf16)acc;
    if (a == 0) {
        float s = 0.0f;
#pragma unroll 4
        for (int p = 0; p < 256; ++p) s += kspart[(size_t)p * 256 + b];
        ksum[b] = s;
    }
}

// ---------------------------------------------------------------------------
// K3b: Mt = (kv @ Wo^T)^T, stored transposed (bf16).  1 WG, 16 waves.
// ---------------------------------------------------------------------------
__global__ void k_kvwo(const __bf16* __restrict__ kvb, const __bf16* __restrict__ Wo,
                       __bf16* __restrict__ Mt) {
    const int t    = threadIdx.x;
    const int lane = t & 31, w = t >> 5;
    const int half = lane >> 4, ln = lane & 15;
    const int at = w;  // a-tile per wave

    bf16x16 af[8];
#pragma unroll
    for (int kk = 0; kk < 8; ++kk)
        af[kk] = frag_a(kvb + at * 16 * CDIM + kk * 32, CDIM);
#pragma unroll
    for (int ct = 0; ct < 16; ++ct) {
        f32x8 acc = {0.f,0.f,0.f,0.f,0.f,0.f,0.f,0.f};
#pragma unroll
        for (int kk = 0; kk < 8; ++kk)
            acc = wmma_bf16(af[kk], frag_b(Wo + ct * 16 * CDIM + kk * 32, CDIM), acc);
        bf16x8 o;
#pragma unroll
        for (int j = 0; j < 8; ++j) o[j] = (__bf16)acc[j];
        int c = ct * 16 + ln;
        int abase = at * 16 + half * 8;
        *(bf16x8*)(Mt + c * CDIM + abase) = o;  // Mt[c][a], contiguous in a
    }
}

// ---------------------------------------------------------------------------
// K4: out = (q @ M) * z + bo,  z = 1/max(q . ksum, 1e-6).  128 rows/WG.
// smem: qs bf16[128][264] @0, ostage f32[128][68] @67584, ksums @102400,
//       zr @103424, boS @103936 -> total 104960 B
// ---------------------------------------------------------------------------
__global__ void k_final(const __bf16* __restrict__ q, const __bf16* __restrict__ Mt,
                        const float* __restrict__ ksum, const float* __restrict__ bo,
                        float* __restrict__ out) {
    extern __shared__ char smem[];
    __bf16* qs     = (__bf16*)(smem);
    float*  ostage = (float*)(smem + 67584);
    float*  ksums  = (float*)(smem + 102400);
    float*  zr     = (float*)(smem + 103424);
    float*  boS    = (float*)(smem + 103936);

    const int t    = threadIdx.x;
    const int lane = t & 31, w = t >> 5;
    const int half = lane >> 4, ln = lane & 15;
    const size_t row0 = (size_t)blockIdx.x * 128;

    // stage q tile (bf16, pure copy) into LDS
#ifdef USE_ASYNC_LDS
#pragma unroll
    for (int i = 0; i < 16; ++i) {
        int idx = t + i * 256;
        int r = idx >> 5, g = idx & 31;
        async_copy_b128(q + (row0 + r) * CDIM + g * 8, qs + r * 264 + g * 8);
    }
    async_wait0();
#else
#pragma unroll
    for (int i = 0; i < 16; ++i) {
        int idx = t + i * 256;
        int r = idx >> 5, g = idx & 31;
        bf16x8 v = *(const bf16x8*)(q + (row0 + r) * CDIM + g * 8);
        *(bf16x8*)(qs + r * 264 + g * 8) = v;
    }
#endif
    ksums[t] = ksum[t];
    boS[t]   = bo[t];
    __syncthreads();

    if (t < 128) {  // per-row normalizer
        float acc = 0.0f;
#pragma unroll
        for (int g = 0; g < 32; ++g) {
            bf16x8 qq = *(const bf16x8*)(qs + t * 264 + g * 8);
#pragma unroll
            for (int e = 0; e < 8; ++e) acc += (float)qq[e] * ksums[g * 8 + e];
        }
        zr[t] = 1.0f / fmaxf(acc, 1e-6f);
    }
    __syncthreads();

    const int m = w;
    bf16x16 af[8];
#pragma unroll
    for (int kk = 0; kk < 8; ++kk) af[kk] = frag_a(qs + m * 16 * 264 + kk * 32, 264);

    for (int c = 0; c < 4; ++c) {
#pragma unroll
        for (int nt = 0; nt < 4; ++nt) {
            int n0 = c * 64 + nt * 16;
            f32x8 acc = {0.f,0.f,0.f,0.f,0.f,0.f,0.f,0.f};
#pragma unroll
            for (int kk = 0; kk < 8; ++kk)
                acc = wmma_bf16(af[kk], frag_b(Mt + n0 * CDIM + kk * 32, CDIM), acc);
#pragma unroll
            for (int j = 0; j < 8; ++j)
                ostage[(m * 16 + half * 8 + j) * 68 + nt * 16 + ln] = acc[j];
        }
        __syncthreads();
#pragma unroll
        for (int i = 0; i < 8; ++i) {  // coalesced f32 store with z + bias
            int idx = t + i * 256;
            int r = idx >> 4, c4 = idx & 15;
            float4 v = *(float4*)(ostage + r * 68 + c4 * 4);
            float z = zr[r];
            int col0 = c * 64 + c4 * 4;
            float4 o;
            o.x = v.x * z + boS[col0 + 0];
            o.y = v.y * z + boS[col0 + 1];
            o.z = v.z * z + boS[col0 + 2];
            o.w = v.w * z + boS[col0 + 3];
            *(float4*)(out + (row0 + r) * CDIM + col0) = o;
        }
        __syncthreads();
    }
}

// ---------------------------------------------------------------------------
// Host launcher
// ---------------------------------------------------------------------------
extern "C" void kernel_launch(void* const* d_in, const int* in_sizes, int n_in,
                              void* d_out, int out_size, void* d_ws, size_t ws_size,
                              hipStream_t stream) {
    const float* x  = (const float*)d_in[0];
    const float* Wq = (const float*)d_in[1];
    const float* Wk = (const float*)d_in[2];
    const float* Wv = (const float*)d_in[3];
    const float* Wo = (const float*)d_in[4];
    const float* bo = (const float*)d_in[5];
    float* out = (float*)d_out;

    char* ws = (char*)d_ws;
    // workspace layout (bytes)
    __bf16* WQb    = (__bf16*)(ws + 0);
    __bf16* WKb    = (__bf16*)(ws + 131072);
    __bf16* WVb    = (__bf16*)(ws + 262144);
    __bf16* WOb    = (__bf16*)(ws + 393216);
    __bf16* KVb    = (__bf16*)(ws + 524288);
    __bf16* Mtb    = (__bf16*)(ws + 655360);
    float*  KSUM   = (float*)(ws + 786432);
    __bf16* Qbuf   = (__bf16*)(ws + 1048576);                 // 128 MiB
    float*  KVPART = (float*)(ws + 1048576 + 134217728);      //  64 MiB
    float*  KSPART = (float*)(ws + 1048576 + 134217728 + 67108864);

    const int WC = CDIM * CDIM;  // 65536
    k_cvt<<<dim3(WC / 256), dim3(256), 0, stream>>>(Wq, WQb, WC);
    k_cvt<<<dim3(WC / 256), dim3(256), 0, stream>>>(Wk, WKb, WC);
    k_cvt<<<dim3(WC / 256), dim3(256), 0, stream>>>(Wv, WVb, WC);
    k_cvt<<<dim3(WC / 256), dim3(256), 0, stream>>>(Wo, WOb, WC);

    k_qproj<<<dim3(NROW / 128), dim3(256), 86016, stream>>>(x, WQb, Qbuf);
    k_kv<<<dim3(256), dim3(1024), 209920, stream>>>(x, WKb, WVb, KVPART, KSPART);
    k_reduce<<<dim3(256), dim3(256), 0, stream>>>(KVPART, KSPART, KVb, KSUM);
    k_kvwo<<<dim3(1), dim3(512), 0, stream>>>(KVb, WOb, Mtb);
    k_final<<<dim3(NROW / 128), dim3(256), 104960, stream>>>(Qbuf, Mtb, KSUM, bo, out);
}
